// skipLSTM_36842229465460
// MI455X (gfx1250) — matrix operand order repaired and checked by reading, not compiled, not run on hardware
//
#include <hip/hip_runtime.h>
#include <hip/hip_bf16.h>

// ---------------------------------------------------------------------------
// skip-LSTM for MI455X (gfx1250, wave32, WMMA)
//   B=128, T=1024, D=512, U=512, 4U=2048
//
//  1) convert X -> bf16, W/R -> bf16 K-major (transposed)
//  2) big WMMA GEMM (2x2 tiles/wave, NT stores): Zx[t][b][4U] = X @ W
//  3) persistent recurrent kernel (64 blocks), 2 device-wide barriers/step:
//       phase1: ut finalize (block 0) || async-stage h-tile to LDS ||
//               zbuf = h @ R via v_wmma_f32_16x16x32_bf16 (A from LDS)
//       phase2: LSTM elementwise + skip blend, NT out, wave-reduced dense head
// ---------------------------------------------------------------------------

typedef __bf16 bf16_t;
typedef __attribute__((ext_vector_type(16))) __bf16 v16bf;
typedef __attribute__((ext_vector_type(8)))  __bf16 v8bf;
typedef __attribute__((ext_vector_type(8)))  float  v8f;

#define B_  128
#define T_  1024
#define D_  512
#define U_  512
#define N4_ 2048

// ---- workspace layout (bytes) ----
static constexpr size_t OFF_XBF  = 0;                                   // B*T*D bf16
static constexpr size_t SZ_XBF   = (size_t)B_ * T_ * D_ * 2;
static constexpr size_t OFF_ZX   = OFF_XBF + SZ_XBF;                    // T*B*4U bf16 (time-major)
static constexpr size_t SZ_ZX    = (size_t)T_ * B_ * N4_ * 2;
static constexpr size_t OFF_WT   = OFF_ZX + SZ_ZX;                      // W^T  [4U][D] bf16
static constexpr size_t SZ_WT    = (size_t)N4_ * D_ * 2;
static constexpr size_t OFF_RKT  = OFF_WT + SZ_WT;                      // R^T  [4U][U] bf16
static constexpr size_t SZ_RKT   = (size_t)N4_ * U_ * 2;
static constexpr size_t OFF_H    = OFF_RKT + SZ_RKT;                    // h carry f32 [B][U]
static constexpr size_t OFF_C    = OFF_H  + (size_t)B_ * U_ * 4;        // c carry f32
static constexpr size_t OFF_HBF  = OFF_C  + (size_t)B_ * U_ * 4;        // h carry bf16 (A-matrix)
static constexpr size_t OFF_ZBUF = OFF_HBF + (size_t)B_ * U_ * 2;       // h@R f32 [B][4U]
static constexpr size_t OFF_UT   = OFF_ZBUF + (size_t)B_ * N4_ * 4;     // ut f32 [B]
static constexpr size_t OFF_ACC  = OFF_UT  + 512;                       // dense-head acc f32 [B]
static constexpr size_t OFF_BAR  = OFF_ACC + 512;                       // barrier {count, generation}

// ---------------------------------------------------------------------------
// WMMA fragment loaders per CDNA5 ISA 7.12.2 (wave32).
// A 16x32 bf16 row-major:  lanes 0-15: M=lane,    K[0..7] then K[16..23]
//                          lanes16-31: M=lane-16, K[8..15] then K[24..31]
// B 32x16 bf16 from K-major Bt[n][k]: lanes 0-15: N=lane,    K[0..7] then K[16..23]
//                                     lanes16-31: N=lane-16, K[8..15] then K[24..31]
// ---------------------------------------------------------------------------
__device__ __forceinline__ v16bf pack16(v8bf lo, v8bf hi) {
  v16bf r;
#pragma unroll
  for (int i = 0; i < 8; ++i) { r[i] = lo[i]; r[i + 8] = hi[i]; }
  return r;
}

__device__ __forceinline__ v16bf load_a16x32(const bf16_t* __restrict__ A, int lda,
                                             int mbase, int kbase, int lane) {
  int m  = mbase + (lane & 15);
  int k0 = kbase + ((lane >> 4) << 3);
  const bf16_t* p = A + (size_t)m * lda + k0;
  return pack16(*(const v8bf*)(p), *(const v8bf*)(p + 16));
}

// A fragment from an LDS-staged 16x512 row-major tile
__device__ __forceinline__ v16bf lds_a16x32(const bf16_t* sA, int kbase, int lane) {
  int m  = lane & 15;
  int k0 = kbase + ((lane >> 4) << 3);
  const bf16_t* p = sA + m * 512 + k0;
  return pack16(*(const v8bf*)(p), *(const v8bf*)(p + 16));
}

__device__ __forceinline__ v16bf load_b32x16(const bf16_t* __restrict__ Bt, int ldb,
                                             int nbase, int kbase, int lane) {
  int n  = nbase + (lane & 15);
  int k0 = kbase + ((lane >> 4) << 3);
  const bf16_t* p = Bt + (size_t)n * ldb + k0;
  return pack16(*(const v8bf*)(p), *(const v8bf*)(p + 16));
}

__device__ __forceinline__ v8f wmma_bf16(v16bf a, v16bf b, v8f c) {
  return __builtin_amdgcn_wmma_f32_16x16x32_bf16(false, a, false, b, (short)0, c,
                                                 false, false);
}

__device__ __forceinline__ float sigmoidf_(float x) {
  return 1.0f / (1.0f + __expf(-x));
}

// ---------------------------------------------------------------------------
// conversion / init kernels
// ---------------------------------------------------------------------------
__global__ __launch_bounds__(256) void k_cvt_x(const float* __restrict__ x,
                                               bf16_t* __restrict__ xb) {
  size_t i = ((size_t)blockIdx.x * blockDim.x + threadIdx.x) * 4;
  float4 v = *(const float4*)(x + i);
  xb[i + 0] = (bf16_t)v.x;
  xb[i + 1] = (bf16_t)v.y;
  xb[i + 2] = (bf16_t)v.z;
  xb[i + 3] = (bf16_t)v.w;
}

// in: [K=512][N=2048] row-major f32  ->  out: [N][K] bf16 (K-major)
__global__ __launch_bounds__(256) void k_cvt_wt(const float* __restrict__ in,
                                                bf16_t* __restrict__ out) {
  int i = blockIdx.x * blockDim.x + threadIdx.x;   // over 512*2048
  int k = i >> 11;
  int n = i & (N4_ - 1);
  out[(size_t)n * D_ + k] = (bf16_t)in[i];
}

__global__ __launch_bounds__(256) void k_init(float* h, float* c, bf16_t* hbf,
                                              float* ut, float* acc, unsigned* bar) {
  int i = blockIdx.x * blockDim.x + threadIdx.x;   // 65536 threads
  h[i] = 0.0f;
  c[i] = 0.0f;
  hbf[i] = (bf16_t)0.0f;
  if (i < B_) { ut[i] = 1.0f; acc[i] = 0.0f; }
  if (i == 0) { bar[0] = 0u; bar[1] = 0u; }
}

// ---------------------------------------------------------------------------
// big time-parallel GEMM: Zx[t][b][n] = sum_k X[b*T+t][k] * W[k][n]
// 2x2 16x16 tiles per wave (M=32, N=32): 4 WMMAs per k-iter, shared fragments.
// ---------------------------------------------------------------------------
__global__ __launch_bounds__(256) void k_xgemm(const bf16_t* __restrict__ Xb,
                                               const bf16_t* __restrict__ Wt,
                                               bf16_t* __restrict__ Zx) {
  int wave  = blockIdx.x * (blockDim.x >> 5) + (threadIdx.x >> 5); // 0..262143
  int lane  = threadIdx.x & 31;
  int mpair = wave >> 6;          // 4096 pairs of row tiles (B*T/32)
  int npair = wave & 63;          // 64 pairs of col tiles (2048/32)
  int mbase = mpair << 5;
  int n0    = npair << 5;

  v8f c00 = {}, c01 = {}, c10 = {}, c11 = {};
#pragma unroll 4
  for (int kk = 0; kk < D_; kk += 32) {
    __builtin_prefetch(Xb + (size_t)mbase * D_ + kk + 64, 0, 1);
    __builtin_prefetch(Wt + (size_t)n0 * D_ + kk + 64, 0, 1);
    v16bf a0 = load_a16x32(Xb, D_, mbase,      kk, lane);
    v16bf a1 = load_a16x32(Xb, D_, mbase + 16, kk, lane);
    v16bf b0 = load_b32x16(Wt, D_, n0,         kk, lane);
    v16bf b1 = load_b32x16(Wt, D_, n0 + 16,    kk, lane);
    c00 = wmma_bf16(a0, b0, c00);
    c01 = wmma_bf16(a0, b1, c01);
    c10 = wmma_bf16(a1, b0, c10);
    c11 = wmma_bf16(a1, b1, c11);
  }

  // C layout: VGPR r -> M = r (+8 for lanes 16-31), N = lane&15.
  // Non-temporal stores: Zx is a 537MB once-written stream; keep X/W in L2.
  int moff = (lane >> 4) << 3;
  int col  = lane & 15;
#pragma unroll
  for (int r = 0; r < 8; ++r) {
    int g0 = mbase + r + moff;       // rows of tile pair (b*T + t order)
    int g1 = g0 + 16;
    size_t base0 = ((size_t)(g0 & (T_ - 1)) * B_ + (g0 >> 10)) * N4_;
    size_t base1 = ((size_t)(g1 & (T_ - 1)) * B_ + (g1 >> 10)) * N4_;
    __builtin_nontemporal_store((bf16_t)c00[r], &Zx[base0 + n0 + col]);
    __builtin_nontemporal_store((bf16_t)c01[r], &Zx[base0 + n0 + 16 + col]);
    __builtin_nontemporal_store((bf16_t)c10[r], &Zx[base1 + n0 + col]);
    __builtin_nontemporal_store((bf16_t)c11[r], &Zx[base1 + n0 + 16 + col]);
  }
}

// ---------------------------------------------------------------------------
// device-wide sense-reversal barrier (64 co-resident blocks)
// ---------------------------------------------------------------------------
__device__ __forceinline__ void grid_sync(unsigned* bar, unsigned* gen, unsigned nblk) {
  __syncthreads();
  if (threadIdx.x == 0) {
    __threadfence();
    unsigned g = __hip_atomic_load(gen, __ATOMIC_RELAXED, __HIP_MEMORY_SCOPE_AGENT);
    unsigned a = __hip_atomic_fetch_add(bar, 1u, __ATOMIC_ACQ_REL, __HIP_MEMORY_SCOPE_AGENT);
    if (a == nblk - 1u) {
      __hip_atomic_store(bar, 0u, __ATOMIC_RELAXED, __HIP_MEMORY_SCOPE_AGENT);
      __hip_atomic_store(gen, g + 1u, __ATOMIC_RELEASE, __HIP_MEMORY_SCOPE_AGENT);
    } else {
      while (__hip_atomic_load(gen, __ATOMIC_ACQUIRE, __HIP_MEMORY_SCOPE_AGENT) == g)
        __builtin_amdgcn_s_sleep(1);
    }
  }
  __syncthreads();
}

// ---------------------------------------------------------------------------
// persistent recurrent kernel: 64 blocks x 256 threads (512 waves),
// 2 device-wide barriers per timestep.
// Each block owns one 16-row h-tile (all 8 waves share it) -> async-stage to LDS.
// ---------------------------------------------------------------------------
__global__ __launch_bounds__(256) void k_recurrent(
    const bf16_t* __restrict__ Zx, const bf16_t* __restrict__ RKt,
    const float* __restrict__ bias, const float* __restrict__ dk,
    const float* __restrict__ db, float* __restrict__ h, float* __restrict__ c,
    bf16_t* __restrict__ hbf, float* __restrict__ zbuf, float* __restrict__ ut,
    float* __restrict__ acc, unsigned* __restrict__ bar, unsigned* __restrict__ gen,
    float* __restrict__ out) {
  __shared__ __align__(16) bf16_t sA[16 * 512];   // 16KB staged h-tile

  const int tid  = threadIdx.x;
  const int lane = tid & 31;
  const int warp = tid >> 5;
  const int wave = blockIdx.x * 8 + warp;        // 0..511
  const int mtile = wave >> 6;                   // 8 row tiles (B/16); same for whole block
  const int npair = wave & 63;                   // 64 col-tile pairs (2048/32)
  const int mbase = mtile << 4;
  const int n0    = npair << 5;
  const unsigned nblk = gridDim.x;

  // async-stage source/dest for this wave's 2KB slice of the 16KB h-tile
  const char* hbf_tile = (const char*)(hbf + (size_t)mbase * U_);
  const unsigned sA_base = (unsigned)(uintptr_t)(&sA[0]);

  // elementwise mapping: thread -> (batch row eb, 4 consecutive units at eu)
  const int gtid = blockIdx.x * 256 + tid;       // 0..16383
  const int eb   = gtid >> 7;
  const int eu   = (gtid & 127) << 2;
  float dkh[4], dkc[4];
#pragma unroll
  for (int j = 0; j < 4; ++j) {
    dkh[j] = dk[eu + j];
    dkc[j] = dk[U_ + eu + j];
  }
  const float dbias = db[0];

  const int moff = (lane >> 4) << 3;
  const int col  = lane & 15;

  for (int t = 0; t < T_; ++t) {
    // ---- phase 1 -----------------------------------------------------------
    // (a) async-copy the block's 16x512 bf16 h-tile into LDS (ASYNCcnt path)
#pragma unroll
    for (int i = 0; i < 4; ++i) {
      unsigned off = (unsigned)(warp * 2048 + i * 512 + lane * 16);
      unsigned long long ga = (unsigned long long)(hbf_tile + off);
      unsigned ldsa = sA_base + off;
      asm volatile("global_load_async_to_lds_b128 %0, %1, off"
                   :: "v"(ldsa), "v"(ga) : "memory");
    }
    // (b) overlapped: skip-gate finalize for step t-1 (block 0, 128 lanes)
    if (blockIdx.x == 0 && tid < B_ && t > 0) {
      float a_  = acc[tid];
      float u_  = ut[tid];
      float ugg = rintf(u_);
      float delta = sigmoidf_(a_ + dbias);
      ut[tid]  = ugg * delta + (1.0f - ugg) * (u_ + fminf(delta, 1.0f - u_));
      acc[tid] = 0.0f;
    }
    asm volatile("s_wait_asynccnt 0" ::: "memory");
    __syncthreads();

    // (c) zbuf = h @ R, A fragments from LDS, B streamed from L2-resident R^T
    v8f c0 = {}, c1 = {};
#pragma unroll 4
    for (int kk = 0; kk < U_; kk += 32) {
      __builtin_prefetch(RKt + (size_t)n0 * U_ + kk + 64, 0, 1);
      v16bf a  = lds_a16x32(sA, kk, lane);
      v16bf b0 = load_b32x16(RKt, U_, n0,      kk, lane);
      v16bf b1 = load_b32x16(RKt, U_, n0 + 16, kk, lane);
      c0 = wmma_bf16(a, b0, c0);
      c1 = wmma_bf16(a, b1, c1);
    }
#pragma unroll
    for (int r = 0; r < 8; ++r) {
      int row = mbase + r + moff;
      zbuf[(size_t)row * N4_ + n0 + col]      = c0[r];
      zbuf[(size_t)row * N4_ + n0 + 16 + col] = c1[r];
    }
    grid_sync(bar, gen, nblk);

    // ---- phase 2: LSTM elementwise + skip blend + dense-head partials ------
    const bf16_t* Zrow = Zx + (size_t)t * B_ * N4_ + (size_t)eb * N4_;
    const float*  zb   = zbuf + (size_t)eb * N4_;
    const float   ug   = rintf(ut[eb]);
    float partial = 0.0f;
#pragma unroll
    for (int j = 0; j < 4; ++j) {
      int u = eu + j;
      float zi = (float)__builtin_nontemporal_load(&Zrow[u])          + zb[u]          + bias[u];
      float zf = (float)__builtin_nontemporal_load(&Zrow[U_ + u])     + zb[U_ + u]     + bias[U_ + u];
      float zg = (float)__builtin_nontemporal_load(&Zrow[2 * U_ + u]) + zb[2 * U_ + u] + bias[2 * U_ + u];
      float zo = (float)__builtin_nontemporal_load(&Zrow[3 * U_ + u]) + zb[3 * U_ + u] + bias[3 * U_ + u];
      size_t hi = (size_t)eb * U_ + u;
      float cc = c[hi];
      float hh = h[hi];
      float cn = sigmoidf_(zf) * cc + sigmoidf_(zi) * tanhf(zg);
      float hn = sigmoidf_(zo) * tanhf(cn);
      __builtin_nontemporal_store(hn, &out[((size_t)eb * T_ + t) * U_ + u]); // out = h_n
      float hb = ug * hh + (1.0f - ug) * hn;
      float cb = ug * cc + (1.0f - ug) * cn;
      h[hi]   = hb;
      c[hi]   = cb;
      hbf[hi] = (bf16_t)hb;
      partial += hb * dkh[j] + cb * dkc[j];
    }
    // wave32 reduce -> one f32 atomic per wave (4 waves per batch row)
#pragma unroll
    for (int off = 16; off > 0; off >>= 1) partial += __shfl_down(partial, off, 32);
    if (lane == 0) atomicAdd(&acc[eb], partial);
    grid_sync(bar, gen, nblk);
  }
}

// ---------------------------------------------------------------------------
extern "C" void kernel_launch(void* const* d_in, const int* in_sizes, int n_in,
                              void* d_out, int out_size, void* d_ws, size_t ws_size,
                              hipStream_t stream) {
  (void)in_sizes; (void)n_in; (void)out_size; (void)ws_size;
  const float* x    = (const float*)d_in[0];   // (B,T,D)
  const float* W    = (const float*)d_in[1];   // (D,4U)
  const float* RK   = (const float*)d_in[2];   // (U,4U)
  const float* bias = (const float*)d_in[3];   // (4U,)
  const float* dk   = (const float*)d_in[4];   // (2U,1)
  const float* db   = (const float*)d_in[5];   // (1,)
  float* out = (float*)d_out;                  // (B,T,U)

  char* ws = (char*)d_ws;
  bf16_t*   Xb   = (bf16_t*)(ws + OFF_XBF);
  bf16_t*   Zx   = (bf16_t*)(ws + OFF_ZX);
  bf16_t*   Wt   = (bf16_t*)(ws + OFF_WT);
  bf16_t*   RKt  = (bf16_t*)(ws + OFF_RKT);
  float*    h    = (float*)(ws + OFF_H);
  float*    c    = (float*)(ws + OFF_C);
  bf16_t*   hbf  = (bf16_t*)(ws + OFF_HBF);
  float*    zbuf = (float*)(ws + OFF_ZBUF);
  float*    ut   = (float*)(ws + OFF_UT);
  float*    acc  = (float*)(ws + OFF_ACC);
  unsigned* bar  = (unsigned*)(ws + OFF_BAR);

  // 1) precision conversion / weight transposes
  k_cvt_x<<<(B_ * T_ * D_) / (256 * 4), 256, 0, stream>>>(x, Xb);
  k_cvt_wt<<<(D_ * N4_) / 256, 256, 0, stream>>>(W, Wt);
  k_cvt_wt<<<(U_ * N4_) / 256, 256, 0, stream>>>(RK, RKt);
  k_init<<<256, 256, 0, stream>>>(h, c, hbf, ut, acc, bar);

  // 2) time-parallel input GEMM (half the FLOPs, fully parallel, 2x2 tiles)
  k_xgemm<<<(B_ * T_ / 32) * (N4_ / 32) / 8, 256, 0, stream>>>(Xb, Wt, Zx);

  // 3) persistent recurrent sweep over T, 2 device-wide barriers per step
  k_recurrent<<<64, 256, 0, stream>>>(Zx, RKt, bias, dk, db, h, c, hbf, zbuf, ut,
                                      acc, bar, bar + 1, out);
}
